// SAblock_79654463472153
// MI455X (gfx1250) — matrix operand order repaired and checked by reading, not compiled, not run on hardware
//
#include <hip/hip_runtime.h>
#include <hip/hip_bf16.h>
#include <math.h>

// Problem constants (from reference): B=8, C=256, H=W=64 -> N=4096, C'=32
#define BATCH 8
#define CCH   256
#define NPIX  4096
#define CPD   32

typedef __attribute__((ext_vector_type(16))) __bf16 v16bf;
typedef __attribute__((ext_vector_type(8)))  __bf16 v8bf;
typedef __attribute__((ext_vector_type(8)))  float  v8f;

static __device__ __forceinline__ v8f wmma_bf16(v16bf a, v16bf b, v8f c) {
  // D = A(16x32 bf16) x B(32x16 bf16) + C(16x16 f32)
  return __builtin_amdgcn_wmma_f32_16x16x32_bf16(false, a, false, b, (short)0, c,
                                                 false, false);
}

// -------------------------------------------------------------------------
// Kernel 1: QKV projection.  q,k stored [b][n][c'] bf16 (position-major,
// channels contiguous -> matches WMMA A/B fragment loads); v stored
// transposed [b][c'][n] bf16 (for the V^T B-fragment: k=j contiguous).
// -------------------------------------------------------------------------
__global__ __launch_bounds__(256) void qkv_proj_kernel(
    const float* __restrict__ x,
    const float* __restrict__ Wq, const float* __restrict__ bq,
    const float* __restrict__ Wk, const float* __restrict__ bk,
    const float* __restrict__ Wv, const float* __restrict__ bv,
    __bf16* __restrict__ qmat, __bf16* __restrict__ kmat,
    __bf16* __restrict__ vt)
{
  __shared__ float w_s[CPD * CCH];   // 32 KB, one W matrix at a time
  const int tid = threadIdx.x;
  const int b   = blockIdx.y;
  const int n   = blockIdx.x * 256 + tid;

  const float* Ws[3] = {Wq, Wk, Wv};
  const float* bs[3] = {bq, bk, bv};

  for (int mtx = 0; mtx < 3; ++mtx) {
    __syncthreads();   // protect w_s reuse across passes
    for (int i = tid; i < CPD * CCH; i += 256) w_s[i] = Ws[mtx][i];
    __syncthreads();

    float acc[CPD];
    #pragma unroll
    for (int o = 0; o < CPD; ++o) acc[o] = bs[mtx][o];

    const float* xp = x + (size_t)b * CCH * NPIX + n;
    for (int c = 0; c < CCH; ++c) {
      const float xv = xp[(size_t)c * NPIX];   // coalesced across lanes
      #pragma unroll
      for (int o = 0; o < CPD; ++o)
        acc[o] = fmaf(w_s[o * CCH + c], xv, acc[o]);   // w_s broadcast
    }

    if (mtx < 2) {
      __bf16* dst = (mtx == 0 ? qmat : kmat) + ((size_t)b * NPIX + n) * CPD;
      #pragma unroll
      for (int g = 0; g < 4; ++g) {
        v8bf p;
        #pragma unroll
        for (int e = 0; e < 8; ++e) p[e] = (__bf16)acc[g * 8 + e];
        *(v8bf*)(dst + g * 8) = p;   // 16B stores, contiguous per lane
      }
    } else {
      #pragma unroll
      for (int o = 0; o < CPD; ++o)
        vt[((size_t)b * CPD + o) * NPIX + n] = (__bf16)acc[o];  // coalesced in n
    }
  }
}

// -------------------------------------------------------------------------
// Kernel 2: row softmax stats.  For each q-row j: M_j = max_i S[j,i],
// R_j = 1/sum_i exp(S[j,i]-M_j).  S tile via WMMA, online max/rescale.
// Wave handles 16 rows; streams all 4096 columns in chunks of 64.
// -------------------------------------------------------------------------
__global__ __launch_bounds__(256) void softmax_stats_kernel(
    const __bf16* __restrict__ qmat, const __bf16* __restrict__ kmat,
    float* __restrict__ Mbuf, float* __restrict__ Rbuf)
{
  const int tid  = threadIdx.x;
  const int wave = tid >> 5;
  const int lane = tid & 31;
  const int l15  = lane & 15;
  const bool hi  = (lane >= 16);
  const int b    = blockIdx.y;
  const int j0w  = blockIdx.x * 64 + wave * 16;   // q rows of this wave

  // A fragment of Q (rows j): lane<16 -> c in {0..15}; lane>=16 -> +16
  v16bf aq;
  {
    const int row = j0w + l15;
    const int cb  = hi ? 16 : 0;
    const __bf16* qr = qmat + ((size_t)b * NPIX + row) * CPD;
    v8bf lo = *(const v8bf*)(qr + cb);
    v8bf hi8 = *(const v8bf*)(qr + cb + 8);
    #pragma unroll
    for (int e = 0; e < 8; ++e) { aq[e] = lo[e]; aq[e + 8] = hi8[e]; }
  }

  float m[8], l[8];
  #pragma unroll
  for (int r = 0; r < 8; ++r) { m[r] = -1e30f; l[r] = 0.f; }

  for (int i0 = 0; i0 < NPIX; i0 += 64) {
    v8f s[4];
    #pragma unroll
    for (int t = 0; t < 4; ++t) {
      // B fragment of K (cols i): 16 contiguous channels per lane
      const int col = i0 + 16 * t + l15;
      const int kb  = hi ? 16 : 0;
      const v16bf bkf = *(const v16bf*)(kmat + ((size_t)b * NPIX + col) * CPD + kb);
      v8f z;
      #pragma unroll
      for (int e = 0; e < 8; ++e) z[e] = 0.f;
      s[t] = wmma_bf16(aq, bkf, z);
    }
    if (i0 + 64 < NPIX)
      __builtin_prefetch(kmat + ((size_t)b * NPIX + (i0 + 64) + l15) * CPD, 0, 1);

    #pragma unroll
    for (int r = 0; r < 8; ++r) {
      float vm = fmaxf(fmaxf(s[0][r], s[1][r]), fmaxf(s[2][r], s[3][r]));
      #pragma unroll
      for (int off = 8; off >= 1; off >>= 1)
        vm = fmaxf(vm, __shfl_xor(vm, off, 32));   // reduce within 16-lane half
      const float mn    = fmaxf(m[r], vm);
      const float alpha = __expf(m[r] - mn);
      float ps = __expf(s[0][r] - mn) + __expf(s[1][r] - mn) +
                 __expf(s[2][r] - mn) + __expf(s[3][r] - mn);
      #pragma unroll
      for (int off = 8; off >= 1; off >>= 1)
        ps += __shfl_xor(ps, off, 32);
      l[r] = l[r] * alpha + ps;
      m[r] = mn;
    }
  }

  if (l15 == 0) {   // lanes 0 and 16 write rows r and r+8
    #pragma unroll
    for (int r = 0; r < 8; ++r) {
      const int j = j0w + r + (hi ? 8 : 0);
      Mbuf[(size_t)b * NPIX + j] = m[r];
      Rbuf[(size_t)b * NPIX + j] = 1.0f / l[r];
    }
  }
}

// -------------------------------------------------------------------------
// Kernel 3: out[c',i] = sum_j exp(S[j,i]-M_j)*R_j * v[c',j], then fused
// back-projection Wb + bias + residual.  Computes S^T tiles (A=K-frag,
// B=Q-frag), P bounced through LDS into A-fragment layout (D-layout !=
// A-layout per ISA 7.12.2), then O += P x V^T via WMMA.  No rescale needed
// (stats are final).  Wave owns 16 output positions i.
// -------------------------------------------------------------------------
__global__ __launch_bounds__(256) void attn_out_kernel(
    const __bf16* __restrict__ qmat, const __bf16* __restrict__ kmat,
    const __bf16* __restrict__ vt,
    const float* __restrict__ Mbuf, const float* __restrict__ Rbuf,
    const float* __restrict__ Wb, const float* __restrict__ bb,
    const float* __restrict__ x, float* __restrict__ out)
{
  __shared__ __bf16 p_s[4][16 * 64];   // per-wave P tile (16 i x 64 j), 8 KB
  __shared__ float  o_s[64][33];       // O' tile [n_local][c'], padded vs banks

  const int tid  = threadIdx.x;
  const int wave = tid >> 5;
  const int lane = tid & 31;
  const int l15  = lane & 15;
  const bool hi  = (lane >= 16);
  const int b    = blockIdx.y;
  const int n0   = blockIdx.x * 64;
  const int i0   = n0 + wave * 16;     // output (k) positions of this wave

  // A fragment of K^T (rows i, k = channel)
  v16bf akf;
  {
    const int row = i0 + l15;
    const int cb  = hi ? 16 : 0;
    const __bf16* kr = kmat + ((size_t)b * NPIX + row) * CPD;
    v8bf lo = *(const v8bf*)(kr + cb);
    v8bf hi8 = *(const v8bf*)(kr + cb + 8);
    #pragma unroll
    for (int e = 0; e < 8; ++e) { akf[e] = lo[e]; akf[e + 8] = hi8[e]; }
  }

  v8f o0, o1;
  #pragma unroll
  for (int e = 0; e < 8; ++e) { o0[e] = 0.f; o1[e] = 0.f; }

  __bf16* pw = &p_s[wave][0];
  const float* Mb = Mbuf + (size_t)b * NPIX;
  const float* Rb = Rbuf + (size_t)b * NPIX;

  for (int j0 = 0; j0 < NPIX; j0 += 64) {
    // S^T tile: rows i (ours), cols j (q positions)
    v8f s[4];
    float mc[4], rc[4];
    #pragma unroll
    for (int t = 0; t < 4; ++t) {
      const int col = j0 + 16 * t + l15;
      const int kb  = hi ? 16 : 0;
      const v16bf bqf = *(const v16bf*)(qmat + ((size_t)b * NPIX + col) * CPD + kb);
      v8f z;
      #pragma unroll
      for (int e = 0; e < 8; ++e) z[e] = 0.f;
      s[t]  = wmma_bf16(akf, bqf, z);
      mc[t] = Mb[col];
      rc[t] = Rb[col];
    }
    if (j0 + 64 < NPIX) {
      __builtin_prefetch(qmat + ((size_t)b * NPIX + (j0 + 64) + l15) * CPD, 0, 1);
      __builtin_prefetch(vt + ((size_t)b * CPD + l15) * NPIX + (j0 + 64), 0, 1);
    }

    // P[i,j] = exp(S[j,i]-M_j)*R_j ; D layout: row i = r+8*hi, col j = 16t+l15
    #pragma unroll
    for (int r = 0; r < 8; ++r) {
      #pragma unroll
      for (int t = 0; t < 4; ++t) {
        const float p = __expf(s[t][r] - mc[t]) * rc[t];
        pw[(r + (hi ? 8 : 0)) * 64 + 16 * t + l15] = (__bf16)p;
      }
    }

    // O[i,c'] += P(16 x 32j) x V^T(32j x 16c'), two j-subtiles, two c'-tiles
    #pragma unroll
    for (int u = 0; u < 2; ++u) {
      const int base = (hi ? 16 : 0) + 32 * u;
      const __bf16* pr = pw + l15 * 64 + base;        // A-frag layout from LDS
      v8bf plo = *(const v8bf*)pr;
      v8bf phi = *(const v8bf*)(pr + 8);
      v16bf ap;
      #pragma unroll
      for (int e = 0; e < 8; ++e) { ap[e] = plo[e]; ap[e + 8] = phi[e]; }

      #pragma unroll
      for (int ct = 0; ct < 2; ++ct) {
        const int c  = ct * 16 + l15;
        const int jb = j0 + 32 * u + (hi ? 16 : 0);
        const v16bf bvf = *(const v16bf*)(vt + ((size_t)b * CPD + c) * NPIX + jb);
        if (ct == 0) o0 = wmma_bf16(ap, bvf, o0);
        else         o1 = wmma_bf16(ap, bvf, o1);
      }
    }
  }

  // Stage O' tile to LDS
  #pragma unroll
  for (int r = 0; r < 8; ++r) {
    const int nl = wave * 16 + r + (hi ? 8 : 0);
    o_s[nl][l15]      = o0[r];
    o_s[nl][16 + l15] = o1[r];
  }
  __syncthreads();

  // Fused back-projection: back[cc,i] = Wb[cc,:] . O'[i,:] + bb[cc] + x[cc,i]
  const int nl  = tid & 63;        // lane-contiguous -> coalesced stores
  const int cc0 = tid >> 6;        // uniform within a wave -> Wb broadcast
  float orow[CPD];
  #pragma unroll
  for (int o = 0; o < CPD; ++o) orow[o] = o_s[nl][o];   // conflict-free (33 pad)

  const size_t xbase = ((size_t)b * CCH) * NPIX + n0 + nl;
  for (int kk = 0; kk < 64; ++kk) {
    const int cc = cc0 + 4 * kk;
    float acc = bb[cc];
    const float* wr = Wb + cc * CPD;
    #pragma unroll
    for (int o = 0; o < CPD; ++o) acc = fmaf(wr[o], orow[o], acc);
    const size_t xi = xbase + (size_t)cc * NPIX;
    out[xi] = acc + x[xi];
  }
}

// -------------------------------------------------------------------------
extern "C" void kernel_launch(void* const* d_in, const int* in_sizes, int n_in,
                              void* d_out, int out_size, void* d_ws, size_t ws_size,
                              hipStream_t stream) {
  (void)in_sizes; (void)n_in; (void)out_size; (void)ws_size;
  const float* x  = (const float*)d_in[0];
  const float* Wq = (const float*)d_in[1];
  const float* bq = (const float*)d_in[2];
  const float* Wk = (const float*)d_in[3];
  const float* bk = (const float*)d_in[4];
  const float* Wv = (const float*)d_in[5];
  const float* bv = (const float*)d_in[6];
  const float* Wb = (const float*)d_in[7];
  const float* bb = (const float*)d_in[8];
  float* out = (float*)d_out;

  // Workspace layout: qmat/kmat (2 MB each), vt (2 MB), M/R stats (128 KB each)
  const size_t qkv_elems = (size_t)BATCH * NPIX * CPD;
  __bf16* qmat = (__bf16*)d_ws;
  __bf16* kmat = qmat + qkv_elems;
  __bf16* vt   = kmat + qkv_elems;
  float*  Mbuf = (float*)(vt + qkv_elems);
  float*  Rbuf = Mbuf + (size_t)BATCH * NPIX;

  dim3 g1(NPIX / 256, BATCH);
  qkv_proj_kernel<<<g1, 256, 0, stream>>>(x, Wq, bq, Wk, bk, Wv, bv,
                                          qmat, kmat, vt);
  dim3 g2(NPIX / 64, BATCH);
  softmax_stats_kernel<<<g2, 256, 0, stream>>>(qmat, kmat, Mbuf, Rbuf);
  attn_out_kernel<<<g2, 256, 0, stream>>>(qmat, kmat, vt, Mbuf, Rbuf,
                                          Wb, bb, x, out);
}
